// EP2T_446676598907
// MI455X (gfx1250) — compile-verified, matrix operand order checked by reading, not run on hardware
//
#include <hip/hip_runtime.h>
#include <hip/hip_bf16.h>

typedef __attribute__((ext_vector_type(16))) _Float16 v16h;
typedef __attribute__((ext_vector_type(8)))  _Float16 v8h;
typedef __attribute__((ext_vector_type(8)))  float    v8f;

#define HDIM   256
#define WDIM   256
#define HWPIX  65536      // HDIM*WDIM
#define NCH    77         // 64 feat + 3 stacking + 6 vol + 2 counting + 2 timing
#define BATCH  4
#define W2LDSTRIDE 136    // 128 halfs + 8 pad (=68 dwords) -> conflict-free column reads

__device__ __constant__ int d_chbase[5] = {0, 64, 67, 73, 75};
__device__ __constant__ int d_cvn[5]    = {64, 3, 6, 2, 2};

// ---------------------------------------------------------------- t min/max
__global__ void tmm_kernel(const float* __restrict__ flow, float* __restrict__ tmm, int N) {
    __shared__ float smin[256], smax[256];
    const int b = blockIdx.x;
    const float* fB = flow + (size_t)b * N * 4;
    float mn = 1e30f, mx = -1e30f;
    for (int i = threadIdx.x; i < N; i += blockDim.x) {
        float t = fB[(size_t)i * 4 + 2];
        mn = fminf(mn, t); mx = fmaxf(mx, t);
    }
    smin[threadIdx.x] = mn; smax[threadIdx.x] = mx;
    __syncthreads();
    for (int s = 128; s > 0; s >>= 1) {
        if (threadIdx.x < s) {
            smin[threadIdx.x] = fminf(smin[threadIdx.x], smin[threadIdx.x + s]);
            smax[threadIdx.x] = fmaxf(smax[threadIdx.x], smax[threadIdx.x + s]);
        }
        __syncthreads();
    }
    if (threadIdx.x == 0) { tmm[b * 2 + 0] = smin[0]; tmm[b * 2 + 1] = smax[0]; }
}

// ------------------------------------------- fused MLP (WMMA) + scatter-add
// One wave (32 lanes) processes 16 events.  Layer-2 GEMM [16x128]@[128x64] is
// 4 K-blocks x 4 N-tiles of v_wmma_f32_16x16x32_f16; B-tiles are re-streamed
// from LDS each chunk (memory clobber stops LICM from hoisting -> no spills).
__global__ void __launch_bounds__(256, 4)
mlp_scatter(const float* __restrict__ flow,
            const float* __restrict__ W1, const float* __restrict__ b1,
            const float* __restrict__ W2, const float* __restrict__ b2,
            float* __restrict__ acc, const float* __restrict__ tmm, int N)
{
    __shared__ float sW1[4 * 128];
    __shared__ float sb1[128];
    __shared__ float sb2[64];
    __shared__ _Float16 sW2h[64 * W2LDSTRIDE];   // W2 column-major as f16, padded
    __shared__ float sw[8 * 16];                 // per-wave event weights w
    __shared__ float ssum_b2;

    const int b   = blockIdx.y;
    const int tid = threadIdx.x;
    for (int i = tid; i < 512; i += blockDim.x) sW1[i] = W1[i];
    for (int i = tid; i < 128; i += blockDim.x) sb1[i] = b1[i];
    for (int i = tid; i <  64; i += blockDim.x) sb2[i] = b2[i];
    for (int i = tid; i < 128 * 64; i += blockDim.x) {
        const int k = i >> 6, c = i & 63;
        sW2h[c * W2LDSTRIDE + k] = (_Float16)W2[k * 64 + c];
    }
    if (tid == 0) { float s = 0.f; for (int c = 0; c < 64; ++c) s += b2[c]; ssum_b2 = s; }
    __syncthreads();

    const int lane = tid & 31;
    const int wid  = tid >> 5;
    const int half = lane >> 4;           // 0: K 0..7/16..23, 1: K 8..15/24..31
    const int m16  = lane & 15;           // A-matrix row / B-matrix column
    const float sumb2 = ssum_b2;

    const float tmin = tmm[b * 2 + 0];
    const float tinv = 1.0f / (tmm[b * 2 + 1] - tmin);

    const int chunks        = (N + 15) >> 4;
    const int wavesPerBatch = gridDim.x * (blockDim.x >> 5);
    const int gw            = blockIdx.x * (blockDim.x >> 5) + wid;
    const float* flowB = flow + (size_t)b * N * 4;
    float*       accB  = acc  + (size_t)b * HWPIX * NCH;

    for (int ck = gw; ck < chunks; ck += wavesPerBatch) {
        // Stop LICM from hoisting the chunk-invariant LDS reads (W1/W2 tiles)
        // into long-lived registers: re-reading LDS per chunk is far cheaper
        // than the scratch spills a hoist would force.
        asm volatile("" ::: "memory");

        const int base = ck << 4;
        const int e    = base + m16;
        const int ec   = e < N ? e : N - 1;
        const float4 ev = *(const float4*)(flowB + (size_t)ec * 4);
        const float fx  = ev.x * (1.0f / (WDIM - 1));
        const float fy  = ev.y;
        const float ftn = (ev.z - tmin) * tinv;
        const float fp  = ev.w;
        const int pixaddr = (((int)ev.y) * WDIM + (int)ev.x) * NCH;

        // Layer 1: each lane computes the 64 hidden units its A-fragment needs
        // (wave-uniform sW1 addresses per half -> LDS broadcast reads).
        v16h A[4];
#pragma unroll
        for (int kb = 0; kb < 4; ++kb) {
            const int koff = kb * 32 + half * 8;
#pragma unroll
            for (int i = 0; i < 16; ++i) {
                const int j = koff + (i < 8 ? i : 8 + i);   // K 0..7 then 16..23 per half
                float h = fx * sW1[j] + fy * sW1[128 + j] + ftn * sW1[256 + j]
                        + fp * sW1[384 + j] + sb1[j];
                A[kb][i] = (_Float16)(h > 0.f ? h : 0.f);
            }
        }

        // Pre-exchange scatter addresses: lane needs pixel of event (r | r+8).
        int pa8[8];
#pragma unroll
        for (int r = 0; r < 8; ++r) pa8[r] = __shfl(pixaddr, half ? (r + 8) : r, 32);

        // Layer 2: one channel-tile at a time (low live-register footprint).
        v8f rowsum = {};
#pragma unroll
        for (int ct = 0; ct < 4; ++ct) {
            const int c = ct * 16 + m16;
            const _Float16* colp = &sW2h[c * W2LDSTRIDE];
            v8f D = {};
#pragma unroll
            for (int kb = 0; kb < 4; ++kb) {
                const int koff = kb * 32 + half * 8;
                const v8h lo = *(const v8h*)(colp + koff);
                const v8h hi = *(const v8h*)(colp + koff + 16);
                v16h Bv;
#pragma unroll
                for (int i = 0; i < 8; ++i) { Bv[i] = lo[i]; Bv[8 + i] = hi[i]; }
                D = __builtin_amdgcn_wmma_f32_16x16x32_f16(
                        false, A[kb], false, Bv, (short)0, D, false, false);
            }
            rowsum = rowsum + D;
            const float bc = sb2[c];
#pragma unroll
            for (int r = 0; r < 8; ++r) {
                const int m = half ? (r + 8) : r;
                if (base + m < N)
                    atomicAdd(accB + pa8[r] + c, D[r] + bc);   // feat_vol channel c
            }
        }

        // w = mean over 64 channels; xor masks 1/2/4/8 stay inside each half.
#pragma unroll
        for (int r = 0; r < 8; ++r) {
            float s = rowsum[r];
            s += __shfl_xor(s, 1, 32);
            s += __shfl_xor(s, 2, 32);
            s += __shfl_xor(s, 4, 32);
            s += __shfl_xor(s, 8, 32);
            const float wv = (s + sumb2) * (1.0f / 64.0f);
            if (lane == 0)  sw[wid * 16 + r]     = wv;   // event r
            if (lane == 16) sw[wid * 16 + 8 + r] = wv;   // event r+8
        }

        // Scalar volumes: lanes 0..15, one event each.
        if (lane < 16 && (base + lane) < N) {
            const float w    = sw[wid * 16 + lane];
            const float tn01 = ftn;
            const float ts   = tn01 * 2.0f;          // * (N_TBINS-1)
            const float tl   = floorf(ts);
            const float wu   = ts - tl;
            const int   tli  = (int)tl;
            const int   tui  = tli + 1;
            const int   off  = fp > 0.f ? 0 : 3;
            const int   pol  = fp > 0.f ? 0 : 1;
            float* pxl = accB + pixaddr;
            atomicAdd(pxl + 67 + tli + off, (1.0f - wu) * w);          // vol lower bin
            if (tui < 3) atomicAdd(pxl + 67 + tui + off, wu * w);      // vol upper bin
            int sbin = (int)(tn01 * 3.0f);
            sbin = sbin < 0 ? 0 : (sbin > 2 ? 2 : sbin);
            atomicAdd(pxl + 64 + sbin, w);                             // stacking
            atomicAdd(pxl + 73 + pol, w);                              // counting
            atomicAdd(pxl + 75 + pol, tn01 * w);                       // timing
        }
    }
}

// ------------------------------------------------------- nonzero counts m
__global__ void count_nz(const float* __restrict__ acc, unsigned* __restrict__ mcnt, size_t total) {
    __shared__ unsigned cnt[20];
    if (threadIdx.x < 20) cnt[threadIdx.x] = 0;
    __syncthreads();
    const size_t stride = (size_t)gridDim.x * blockDim.x;
    for (size_t i = (size_t)blockIdx.x * blockDim.x + threadIdx.x; i < total; i += stride) {
        const float v = acc[i];
        if (v != 0.f) {
            const int ch = (int)(i % NCH);
            const int b  = (int)(i / ((size_t)HWPIX * NCH));
            const int vv = ch < 64 ? 0 : ch < 67 ? 1 : ch < 73 ? 2 : ch < 75 ? 3 : 4;
            atomicAdd(&cnt[b * 5 + vv], 1u);
        }
    }
    __syncthreads();
    if (threadIdx.x < 20 && cnt[threadIdx.x]) atomicAdd(&mcnt[threadIdx.x], cnt[threadIdx.x]);
}

// --------------------------------------------------- select-state init
__global__ void init_sel(const unsigned* __restrict__ mcnt,
                         unsigned* __restrict__ selPrefix, unsigned* __restrict__ selK) {
    const int s = threadIdx.x;
    if (s < 40) {
        const unsigned m = mcnt[s >> 1];
        unsigned k = (unsigned)(((s & 1) ? 0.9f : 0.1f) * (float)m);
        if (k < 1) k = 1;
        selPrefix[s] = 0u;
        selK[s] = k;                       // 1-based remaining rank
    }
}

// ---------------------------------------- MSB radix-select: histogram pass
__global__ void radix_hist(const float* __restrict__ acc, const unsigned* __restrict__ selPrefix,
                           unsigned* __restrict__ hist, int round) {
    __shared__ unsigned lh[256];
    const int sel = blockIdx.y;
    const int b   = sel / 10;
    const int v   = (sel % 10) >> 1;
    lh[threadIdx.x] = 0;
    __syncthreads();
    const unsigned prefix = selPrefix[sel];
    const int shift = 24 - 8 * round;
    const int cb = d_chbase[v];
    const size_t total = (size_t)d_cvn[v] * HWPIX;
    const float* accB = acc + (size_t)b * HWPIX * NCH;
    const size_t stride = (size_t)gridDim.x * blockDim.x;
    for (size_t i = (size_t)blockIdx.x * blockDim.x + threadIdx.x; i < total; i += stride) {
        const int c   = (int)(i >> 16);
        const int pix = (int)(i & 65535u);
        const float val = accB[(size_t)pix * NCH + cb + c];
        if (val != 0.f) {
            const unsigned u   = __float_as_uint(val);
            const unsigned key = (u & 0x80000000u) ? ~u : (u | 0x80000000u);
            if (round == 0 || (key >> (shift + 8)) == prefix)
                atomicAdd(&lh[(key >> shift) & 255u], 1u);
        }
    }
    __syncthreads();
    if (lh[threadIdx.x]) atomicAdd(&hist[sel * 256 + threadIdx.x], lh[threadIdx.x]);
}

// ------------------------------------------ radix-select: scan + refine
__global__ void radix_scan(unsigned* __restrict__ hist, unsigned* __restrict__ selPrefix,
                           unsigned* __restrict__ selK, float* __restrict__ bounds, int round) {
    __shared__ unsigned sh[256];
    const int sel = blockIdx.x;
    const int t   = threadIdx.x;
    const unsigned cnt = hist[sel * 256 + t];
    hist[sel * 256 + t] = 0;                 // reset for next round
    sh[t] = cnt;
    __syncthreads();
    for (int off = 1; off < 256; off <<= 1) {
        const unsigned add = (t >= off) ? sh[t - off] : 0u;
        __syncthreads();
        sh[t] += add;
        __syncthreads();
    }
    const unsigned incl = sh[t];
    const unsigned excl = incl - cnt;
    const unsigned k    = selK[sel];
    if (k > excl && k <= incl) {             // exactly one winning digit
        const unsigned np = (selPrefix[sel] << 8) | (unsigned)t;
        if (round == 3) {
            const unsigned u = (np & 0x80000000u) ? (np ^ 0x80000000u) : ~np;
            bounds[sel] = __uint_as_float(u);
        } else {
            selPrefix[sel] = np;
            selK[sel]      = k - excl;
        }
    }
}

// --------------------------------------------------------------- max_val
__global__ void compute_maxval(const float* __restrict__ bounds, float* __restrict__ maxval) {
    const int i = threadIdx.x;
    if (i < 20) {
        const float lower = bounds[2 * i], upper = bounds[2 * i + 1];
        const float mv = fmaxf(fabsf(lower), upper);
        maxval[i] = mv > 0.f ? mv : 1.0f;
    }
}

// ------------------------------------------------------------- finalize
__global__ void finalize(const float* __restrict__ acc, const unsigned* __restrict__ mcnt,
                         const float* __restrict__ maxval, float* __restrict__ out, size_t total) {
    const size_t stride = (size_t)gridDim.x * blockDim.x;
    for (size_t i = (size_t)blockIdx.x * blockDim.x + threadIdx.x; i < total; i += stride) {
        const float v = acc[i];
        const int ch = (int)(i % NCH);
        const int b  = (int)(i / ((size_t)HWPIX * NCH));
        const int vv = ch < 64 ? 0 : ch < 67 ? 1 : ch < 73 ? 2 : ch < 75 ? 3 : 4;
        const int s  = b * 5 + vv;
        if (mcnt[s] > 0) {
            const float mv = maxval[s];
            const float c  = v > mv ? mv : (v < -mv ? -mv : v);
            out[i] = c / mv;
        } else {
            out[i] = v;
        }
    }
}

// ---------------------------------------------------------------- launch
extern "C" void kernel_launch(void* const* d_in, const int* in_sizes, int n_in,
                              void* d_out, int out_size, void* d_ws, size_t ws_size,
                              hipStream_t stream) {
    const float* flow = (const float*)d_in[0];
    const float* W1   = (const float*)d_in[1];
    const float* b1   = (const float*)d_in[2];
    const float* W2   = (const float*)d_in[3];
    const float* b2   = (const float*)d_in[4];
    float* out = (float*)d_out;

    const int N = in_sizes[0] / (BATCH * 4);
    const size_t ACC = (size_t)BATCH * HWPIX * NCH;

    float*    acc       = (float*)d_ws;
    float*    tmm       = acc + ACC;                       // [B][2]
    unsigned* mcnt      = (unsigned*)(tmm + 2 * BATCH);    // [20]
    unsigned* selPrefix = mcnt + 20;                       // [40]
    unsigned* selK      = selPrefix + 40;                  // [40]
    unsigned* hist      = selK + 40;                       // [40*256]
    float*    bounds    = (float*)(hist + 40 * 256);       // [40]
    float*    maxval    = bounds + 40;                     // [20]
    const size_t wsWords = (size_t)((float*)(maxval + 20) - (float*)d_ws);

    hipMemsetAsync(d_ws, 0, wsWords * sizeof(float), stream);
    tmm_kernel<<<BATCH, 256, 0, stream>>>(flow, tmm, N);
    mlp_scatter<<<dim3(160, BATCH), 256, 0, stream>>>(flow, W1, b1, W2, b2, acc, tmm, N);
    count_nz<<<1024, 256, 0, stream>>>(acc, mcnt, ACC);
    init_sel<<<1, 64, 0, stream>>>(mcnt, selPrefix, selK);
    for (int r = 0; r < 4; ++r) {
        radix_hist<<<dim3(16, 40), 256, 0, stream>>>(acc, selPrefix, hist, r);
        radix_scan<<<40, 256, 0, stream>>>(hist, selPrefix, selK, bounds, r);
    }
    compute_maxval<<<1, 32, 0, stream>>>(bounds, maxval);
    finalize<<<2048, 256, 0, stream>>>(acc, mcnt, maxval, out, ACC);
}
